// RNNTJoint_41807211659778
// MI455X (gfx1250) — matrix operand order repaired and checked by reading, not compiled, not run on hardware
//
#include <hip/hip_runtime.h>
#include <hip/hip_bf16.h>

// RNN-T joint on MI455X (gfx1250, wave32, WMMA).
// B=8, EH=768, T=512, PH=320, U=128, JH=320, C=34.
//
// Pipeline:
//   1) proj_f16_kernel: E[b,t,:] = enc[b,:,t] @ W_enc^T + b_enc   (f16 out, f32 acc)
//   2) proj_f16_kernel: P[b,u,:] = dec[b,:,u] @ W_pred^T + b_pred (f16 out)
//   3) joint_kernel: per (b, 16t, 16u) tile, A = relu(E+P) built in-register with
//      packed f16 VALU, logits = A @ W_out^T (+b_out), fused log_softmax, f32 out.

typedef __attribute__((ext_vector_type(16))) _Float16 v16h;
typedef __attribute__((ext_vector_type(8)))  _Float16 v8h;
typedef __attribute__((ext_vector_type(4)))  _Float16 v4h;
typedef __attribute__((ext_vector_type(2)))  _Float16 v2h;
typedef __attribute__((ext_vector_type(8)))  float    v8f;

#define BJ  8
#define TT  512
#define UU  128
#define JH  320
#define CC  34
#define EHK 768
#define PHK 320

static __device__ __forceinline__ v8f wmma_f16(v16h a, v16h b, v8f c) {
  // (neg_a, A, neg_b, B, c_mod, C, reuse_a, reuse_b)
  return __builtin_amdgcn_wmma_f32_16x16x32_f16(false, a, false, b, (short)0, c,
                                                false, false);
}

static __device__ __forceinline__ v16h cat8(v8h lo, v8h hi) {
  return __builtin_shufflevector(lo, hi, 0, 1, 2, 3, 4, 5, 6, 7,
                                         8, 9, 10, 11, 12, 13, 14, 15);
}

// ---------------------------------------------------------------------------
// Projection: Y[b,m,j] = sum_k X[b,k,m] * W[j,k] + bias[j], stored as f16.
// Grid: (M/32, JH/64, B). Block: 256 threads = 8 waves.
// Each wave owns one 16x16 output tile (2 M-tiles x 4 N-tiles per WG).
// ---------------------------------------------------------------------------
__global__ __launch_bounds__(256) void proj_f16_kernel(
    const float* __restrict__ X,     // (B, K, M)
    const float* __restrict__ W,     // (JH, K)
    const float* __restrict__ bias,  // (JH)
    _Float16* __restrict__ Y,        // (B, M, JH) f16
    int K, int M) {
  __shared__ _Float16 Alds[32][40];  // [m][k], pitch 40 (80B): conflict-free frags
  __shared__ _Float16 Wlds[64][40];  // [j][k]

  const int tid  = threadIdx.x;
  const int lane = tid & 31;
  const int wv   = tid >> 5;
  const int n    = lane & 15;
  const int g    = lane >> 4;
  const int mt   = wv & 1;   // 0..1  (M sub-tile)
  const int nt   = wv >> 1;  // 0..3  (N sub-tile)

  const int b     = blockIdx.z;
  const int mBase = blockIdx.x * 32;
  const int jBase = blockIdx.y * 64;

  v8f acc = {};

  for (int k0 = 0; k0 < K; k0 += 32) {
    __syncthreads();
    {  // stage X tile: 32k x 32m; each thread packs a k-pair into one b32 store
      const int m  = tid & 31;
      const int kp = tid >> 5;  // 0..7 -> k = 2*kp (+16)
      const float* src = X + ((size_t)b * K + (k0 + 2 * kp)) * M + mBase + m;
#pragma unroll
      for (int p = 0; p < 2; ++p) {
        v2h v;
        v.x = (_Float16)src[(size_t)(16 * p) * M];
        v.y = (_Float16)src[(size_t)(16 * p + 1) * M];
        *(v2h*)&Alds[m][2 * kp + 16 * p] = v;
      }
    }
    {  // stage W tile: 64j x 32k; float2 loads, packed b32 stores
      const int kp = tid & 15;  // k = 2*kp
      const int j  = tid >> 4;  // 0..15, 4 passes
      const float* src = W + (size_t)(jBase + j) * K + k0 + 2 * kp;
#pragma unroll
      for (int p = 0; p < 4; ++p) {
        const float2 f = *(const float2*)(src + (size_t)(16 * p) * K);
        v2h v;
        v.x = (_Float16)f.x;
        v.y = (_Float16)f.y;
        *(v2h*)&Wlds[j + 16 * p][2 * kp] = v;
      }
    }
    __syncthreads();

    // A frag (16x32, 16-bit): lane m = mt*16+n; halves: K = i+8g (i<8), +16 after
    const int m = mt * 16 + n;
    const v8h a1 = *(const v8h*)&Alds[m][8 * g];
    const v8h a2 = *(const v8h*)&Alds[m][8 * g + 16];
    // B frag (32x16, 16-bit): lane col = nt*16+n; halves: K = i+8g (i<8), +16 after
    const int j = nt * 16 + n;
    const v8h b1 = *(const v8h*)&Wlds[j][8 * g];
    const v8h b2 = *(const v8h*)&Wlds[j][8 * g + 16];

    acc = wmma_f16(cat8(a1, a2), cat8(b1, b2), acc);
  }

  // Epilogue: D layout — VGPR v holds row M = v + 8g, col N = lane%16.
  const int jj = jBase + nt * 16 + n;
  const float bj = bias[jj];
#pragma unroll
  for (int v = 0; v < 8; ++v) {
    const int m = mBase + mt * 16 + v + 8 * g;
    Y[((size_t)b * M + m) * JH + jj] = (_Float16)(acc[v] + bj);
  }
}

// ---------------------------------------------------------------------------
// Fused joint: per WG = (b, 16 t, 16 u). A[u,k] = relu(E[t,k]+P[u,k]) built in
// registers (packed f16); logits = A @ W_out^T + b_out; log_softmax; f32 out.
// Grid: (T/16, U/16, B). Block: 256 threads = 8 waves; wave w owns t = 2w, 2w+1.
// ---------------------------------------------------------------------------
__global__ __launch_bounds__(256) void joint_kernel(
    const _Float16* __restrict__ E,   // (B, T, JH) f16 (bias included)
    const _Float16* __restrict__ P,   // (B, U, JH) f16 (bias included)
    const float* __restrict__ Wout,   // (C, JH)
    const float* __restrict__ bout,   // (C)
    float* __restrict__ out) {        // (B, T, U, C)
  __shared__ _Float16 Elds[16][328];  // pitch 328 halves (656B): 16 distinct banks
  __shared__ _Float16 Plds[16][328];
  __shared__ _Float16 Wlds[48][328];  // C padded 34 -> 48 (3 N-tiles)

  const int tid  = threadIdx.x;
  const int lane = tid & 31;
  const int wv   = tid >> 5;
  const int n    = lane & 15;
  const int g    = lane >> 4;

  const int b  = blockIdx.z;
  const int t0 = blockIdx.x * 16;
  const int u0 = blockIdx.y * 16;

  // Stage E/P tiles (b128 copies) and all of W_out (f32->f16 packed) once per WG.
  for (int i = tid; i < 16 * (JH / 8); i += 256) {  // 16 rows x 40 8-half chunks
    const int r = i / 40, k8 = (i % 40) * 8;
    *(uint4*)&Elds[r][k8] =
        *(const uint4*)(E + ((size_t)b * TT + t0 + r) * JH + k8);
    *(uint4*)&Plds[r][k8] =
        *(const uint4*)(P + ((size_t)b * UU + u0 + r) * JH + k8);
  }
  for (int i = tid; i < 48 * (JH / 4); i += 256) {  // 48 rows x 80 4-elem chunks
    const int c = i / 80, k4 = (i % 80) * 4;
    v4h v = {};
    if (c < CC) {
      const float4 f = *(const float4*)(Wout + (size_t)c * JH + k4);
      v.x = (_Float16)f.x;
      v.y = (_Float16)f.y;
      v.z = (_Float16)f.z;
      v.w = (_Float16)f.w;
    }
    *(v4h*)&Wlds[c][k4] = v;
  }
  __syncthreads();

  v8f acc[2][3] = {};  // [t-sub][n-tile]
  const v8h zero = {};

#pragma unroll 1
  for (int k0 = 0; k0 < JH; k0 += 32) {
    // B frags for the 3 class tiles (shared across both t's of this wave)
    v16h bf[3];
#pragma unroll
    for (int ntl = 0; ntl < 3; ++ntl) {
      const _Float16* wrow = &Wlds[ntl * 16 + n][k0 + 8 * g];
      bf[ntl] = cat8(*(const v8h*)wrow, *(const v8h*)(wrow + 16));
    }

    // P frags don't depend on t: hoist
    const v8h p1 = *(const v8h*)&Plds[n][k0 + 8 * g];
    const v8h p2 = *(const v8h*)&Plds[n][k0 + 8 * g + 16];

#pragma unroll
    for (int ti = 0; ti < 2; ++ti) {
      const int t = 2 * wv + ti;
      // A frag rows = u (lane%16): A[u,k] = relu(E[t,k] + P[u,k]), packed f16.
      const v8h e1 = *(const v8h*)&Elds[t][k0 + 8 * g];
      const v8h e2 = *(const v8h*)&Elds[t][k0 + 8 * g + 16];
      const v8h s1 = __builtin_elementwise_max(e1 + p1, zero);  // v_pk_add/max
      const v8h s2 = __builtin_elementwise_max(e2 + p2, zero);
      const v16h a = cat8(s1, s2);
      acc[ti][0] = wmma_f16(a, bf[0], acc[ti][0]);
      acc[ti][1] = wmma_f16(a, bf[1], acc[ti][1]);
      acc[ti][2] = wmma_f16(a, bf[2], acc[ti][2]);
    }
  }

  // Epilogue: bias + log_softmax over C. Lane holds (u = v+8g, c = 16*nt + n).
  const float bo0 = bout[n];
  const float bo1 = bout[16 + n];
  const float bo2 = (n < 2) ? bout[32 + n] : 0.0f;

#pragma unroll
  for (int ti = 0; ti < 2; ++ti) {
    const int tg = t0 + 2 * wv + ti;
#pragma unroll
    for (int v = 0; v < 8; ++v) {
      const int u = u0 + v + 8 * g;
      const float x0 = acc[ti][0][v] + bo0;
      const float x1 = acc[ti][1][v] + bo1;
      const float x2 = acc[ti][2][v] + bo2;

      float mx = fmaxf(x0, x1);
      if (n < 2) mx = fmaxf(mx, x2);
#pragma unroll
      for (int d = 1; d < 16; d <<= 1) mx = fmaxf(mx, __shfl_xor(mx, d, 16));

      float s = __expf(x0 - mx) + __expf(x1 - mx);
      if (n < 2) s += __expf(x2 - mx);
#pragma unroll
      for (int d = 1; d < 16; d <<= 1) s += __shfl_xor(s, d, 16);

      const float lse = mx + __logf(s);
      const size_t base = (((size_t)b * TT + tg) * UU + u) * CC;
      out[base + n]      = x0 - lse;
      out[base + 16 + n] = x1 - lse;
      if (n < 2) out[base + 32 + n] = x2 - lse;
    }
  }
}

// ---------------------------------------------------------------------------
extern "C" void kernel_launch(void* const* d_in, const int* in_sizes, int n_in,
                              void* d_out, int out_size, void* d_ws, size_t ws_size,
                              hipStream_t stream) {
  (void)in_sizes; (void)n_in; (void)out_size; (void)ws_size;
  const float* enc    = (const float*)d_in[0];  // (B, EH, T)
  const float* dec    = (const float*)d_in[1];  // (B, PH, U)
  const float* W_enc  = (const float*)d_in[2];  // (JH, EH)
  const float* b_enc  = (const float*)d_in[3];  // (JH)
  const float* W_pred = (const float*)d_in[4];  // (JH, PH)
  const float* b_pred = (const float*)d_in[5];  // (JH)
  const float* W_out  = (const float*)d_in[6];  // (C, JH)
  const float* b_out  = (const float*)d_in[7];  // (C)
  float* out = (float*)d_out;

  // Workspace: E (B*T*JH f16 = 2.62 MB) then P (B*U*JH f16 = 0.66 MB).
  _Float16* Ew = (_Float16*)d_ws;
  _Float16* Pw = (_Float16*)((char*)d_ws + (size_t)BJ * TT * JH * sizeof(_Float16));

  dim3 blk(256);
  // E projection: M=T=512, K=EH=768
  proj_f16_kernel<<<dim3(TT / 32, JH / 64, BJ), blk, 0, stream>>>(
      enc, W_enc, b_enc, Ew, EHK, TT);
  // P projection: M=U=128, K=PH=320
  proj_f16_kernel<<<dim3(UU / 32, JH / 64, BJ), blk, 0, stream>>>(
      dec, W_pred, b_pred, Pw, PHK, UU);
  // Fused joint + log_softmax
  joint_kernel<<<dim3(TT / 16, UU / 16, BJ), blk, 0, stream>>>(
      Ew, Pw, W_out, b_out, out);
}
